// RNN_OneLayer_78769700209276
// MI455X (gfx1250) — matrix operand order repaired and checked by reading, not compiled, not run on hardware
//
#include <hip/hip_runtime.h>
#include <cmath>

typedef __attribute__((ext_vector_type(16))) _Float16 v16h;
typedef __attribute__((ext_vector_type(8)))  _Float16 v8h;
typedef __attribute__((ext_vector_type(8)))  float    v8f;

#define HIDDEN 4096
#define LENGTH 4096
#define NWG    128
#define TPB    256

// d_ws layout:
//   [0, 32MB)            packed Wh fp16, WMMA B-tile layout
//   [32MB, +16KB)        h ping-pong (2 x 4096 fp16)
//   [32MB+16KB, +8B)     barrier counters {cnt, gen}
static constexpr size_t PACKED_BYTES = (size_t)HIDDEN * (size_t)HIDDEN * 2u; // 32 MB
static constexpr size_t HBUF_OFF     = PACKED_BYTES;
static constexpr size_t CTR_OFF      = PACKED_BYTES + 2u * HIDDEN * 2u;

__global__ void rnn_init_ctrs(unsigned* ctrs) {
    ctrs[0] = 0u;  // cnt
    ctrs[1] = 0u;  // gen
}

// Pack Wh (row-major fp32 [K=4096][J=4096]) into fp16 WMMA B tiles.
// Tile (jt,kt) covers J = jt*16..+15, K = kt*32..+31, stored as 512 halves:
//   element index = lane*16 + e, where lane = n + (kr & 16), e = kr & 15
// (this code's contiguous B k order: lanes 0-15 hold K=0..15, lanes 16-31 hold K=16..31,
//  2 K-values per VGPR, N = lane & 15).
__global__ void rnn_pack_wh(const float* __restrict__ Wh,
                            _Float16* __restrict__ packed) {
    int id = blockIdx.x * blockDim.x + threadIdx.x;   // 0 .. 16M-1
    int j  = id & (HIDDEN - 1);
    int k  = id >> 12;
    int jt = j >> 4, n  = j & 15;
    int kt = k >> 5, kr = k & 31;
    int lane = n + (kr & 16);
    int e    = kr & 15;
    size_t dst = ((size_t)(jt * 128 + kt)) * 512u + (size_t)lane * 16u + (size_t)e;
    packed[dst] = (_Float16)Wh[(size_t)k * HIDDEN + j];
}

__device__ __forceinline__ void grid_barrier(unsigned* cnt, unsigned* gen) {
    __syncthreads();
    if (threadIdx.x == 0) {
        unsigned g = __hip_atomic_load(gen, __ATOMIC_RELAXED, __HIP_MEMORY_SCOPE_AGENT);
        __threadfence();  // make this WG's h stores visible device-wide
        unsigned arrived = __hip_atomic_fetch_add(cnt, 1u, __ATOMIC_ACQ_REL,
                                                  __HIP_MEMORY_SCOPE_AGENT);
        if (arrived == NWG - 1) {
            __hip_atomic_store(cnt, 0u, __ATOMIC_RELAXED, __HIP_MEMORY_SCOPE_AGENT);
            __hip_atomic_fetch_add(gen, 1u, __ATOMIC_RELEASE, __HIP_MEMORY_SCOPE_AGENT);
        } else {
            while (__hip_atomic_load(gen, __ATOMIC_ACQUIRE,
                                     __HIP_MEMORY_SCOPE_AGENT) == g) {
                __builtin_amdgcn_s_sleep(1);
            }
        }
        __threadfence();
    }
    __syncthreads();
}

// Persistent RNN kernel: 128 WGs x 256 threads (8 wave32).
// WG wg owns outputs j = wg*32 .. wg*32+31 (2 column tiles).
// Wave w: column tile (w>>2), K-quarter (w&3) -> 32 K-chunks of 32.
__global__ void __launch_bounds__(TPB, 1)
rnn_persistent(const float* __restrict__ x,
               const float* __restrict__ Wx,
               const float* __restrict__ bias,
               const _Float16* __restrict__ packed,
               _Float16* __restrict__ hbuf,          // 2 * HIDDEN halves
               const float* __restrict__ fc_w,
               const float* __restrict__ fc_b,
               float* __restrict__ out,
               unsigned* __restrict__ ctrs) {
    __shared__ float lds_acc[32];
    __shared__ float red[TPB];

    unsigned* cnt = ctrs + 0;
    unsigned* gen = ctrs + 1;

    const int tid      = threadIdx.x;
    const int wg       = blockIdx.x;
    const int lane     = tid & 31;
    const int wave     = tid >> 5;
    const int tileInWg = wave >> 2;                 // 0..1
    const int kq       = wave & 3;                  // 0..3
    const int jt       = wg * 2 + tileInWg;         // column tile 0..255
    const int laneSel  = (lane & 16);               // 0 or 16 (A-fragment K select)

    // This wave's first packed B tile (already offset by lane).
    const _Float16* tbase =
        packed + ((size_t)jt * 128u + (size_t)kq * 32u) * 512u + (size_t)lane * 16u;

    // h0 = 0 in buffer 0
    if (tid < 32) hbuf[wg * 32 + tid] = (_Float16)0.f;
    grid_barrier(cnt, gen);

    for (int t = 0; t < LENGTH; ++t) {
        const _Float16* hc = hbuf + (t & 1) * HIDDEN;
        _Float16*       hn = hbuf + ((t & 1) ^ 1) * HIDDEN;

        if (tid < 32) lds_acc[tid] = 0.f;
        __syncthreads();

        v8f c = {};
        const _Float16* tb = tbase;
        const _Float16* ha = hc + kq * 1024 + laneSel;
#pragma unroll 4
        for (int i = 0; i < 32; ++i) {
            // Replicated-h A fragment (all 16 rows = h chunk):
            // lanes<16 need K {0..15}, lanes>=16 need K {16..31}.
            v8h lo = *(const v8h*)(ha);
            v8h hi = *(const v8h*)(ha + 8);
            v16h a = __builtin_shufflevector(lo, hi, 0, 1, 2, 3, 4, 5, 6, 7,
                                             8, 9, 10, 11, 12, 13, 14, 15);
            v16h b = *(const v16h*)(tb);
            c = __builtin_amdgcn_wmma_f32_16x16x32_f16(
                false, a, false, b, (short)0, c, false, false);
            ha += 32;
            tb += 512;
        }

        // Row 0 of C: lanes 0..15, c[0] = partial sum for output column (lane).
        if (lane < 16) atomicAdd(&lds_acc[tileInWg * 16 + lane], c[0]);
        __syncthreads();

        if (tid < 32) {
            int j = wg * 32 + tid;
            float v = x[t] * Wx[j] + bias[j] + lds_acc[tid];
            hn[j] = (_Float16)tanhf(v);
        }
        grid_barrier(cnt, gen);
    }

    // Final: sigmoid(fc_w . h_T + fc_b). h_T is in buffer 0 (4096 even steps).
    if (wg == 0) {
        const _Float16* hf = hbuf;
        float p = 0.f;
        for (int i = tid; i < HIDDEN; i += TPB) p += (float)hf[i] * fc_w[i];
        red[tid] = p;
        __syncthreads();
        for (int s = TPB / 2; s > 0; s >>= 1) {
            if (tid < s) red[tid] += red[tid + s];
            __syncthreads();
        }
        if (tid == 0) {
            float sc = red[0] + fc_b[0];
            out[0] = 1.f / (1.f + expf(-sc));
        }
    }
}

extern "C" void kernel_launch(void* const* d_in, const int* in_sizes, int n_in,
                              void* d_out, int out_size, void* d_ws, size_t ws_size,
                              hipStream_t stream) {
    const float* x    = (const float*)d_in[0];  // [4096]
    const float* Wx   = (const float*)d_in[1];  // [4096]
    const float* Wh   = (const float*)d_in[2];  // [4096*4096]
    const float* bias = (const float*)d_in[3];  // [4096]
    const float* fc_w = (const float*)d_in[4];  // [4096]
    const float* fc_b = (const float*)d_in[5];  // [1]
    float* out = (float*)d_out;

    char* ws = (char*)d_ws;
    _Float16* packed = (_Float16*)ws;
    _Float16* hbuf   = (_Float16*)(ws + HBUF_OFF);
    unsigned* ctrs   = (unsigned*)(ws + CTR_OFF);

    rnn_init_ctrs<<<1, 1, 0, stream>>>(ctrs);

    int packBlocks = (HIDDEN * HIDDEN) / TPB;  // 65536
    rnn_pack_wh<<<packBlocks, TPB, 0, stream>>>(Wh, packed);

    rnn_persistent<<<NWG, TPB, 0, stream>>>(x, Wx, bias, packed, hbuf,
                                            fc_w, fc_b, out, ctrs);
}
